// VAE_42021960024338
// MI455X (gfx1250) — matrix operand order repaired and checked by reading, not compiled, not run on hardware
//
#include <hip/hip_runtime.h>
#include <algorithm>

// ---------------------------------------------------------------------------
// CDNA5 / gfx1250 physics-VAE forward pass.
//   * all GEMMs via v_wmma_f32_16x16x32_bf16 (fp32 accum, bf16 operands)
//   * operands read k-contiguously (A: [M][K], B: [N][K]) -> b128 loads only
//   * generic GEMM: barrier-free, LDS-free, software-pipelined (2 stages),
//     32 rows x 64 cols per wave (16 wmma / 24 b128 loads per 64-K slice)
//   * Hamiltonian integrator: persistent per-WG scan; B fragments reused
//     across M-subtiles; W2 backward operand streamed from L2.
// ---------------------------------------------------------------------------

typedef __bf16 bf16;
typedef __attribute__((ext_vector_type(16))) __bf16 v16bf;
typedef __attribute__((ext_vector_type(8)))  float  v8f;

// ---------------- WMMA fragment helpers (wave32, 16x16x32 bf16) -------------
// A fragment: element (m,k) at base[m*stride + k]   (LDS)
__device__ __forceinline__ v16bf frag_a(const bf16* base, int stride) {
  int lane = threadIdx.x & 31;
  int m = lane & 15;
  int k0 = (lane >> 4) * 16;
  const bf16* p = base + m * stride + k0;
  v16bf f;
#pragma unroll
  for (int j = 0; j < 8; ++j) { f[j] = p[j]; f[8 + j] = p[8 + j]; }
  return f;
}

// B fragment from k-contiguous storage: element (k,n) at base[n*stride + k]
__device__ __forceinline__ v16bf frag_bT(const bf16* base, int stride) {
  int lane = threadIdx.x & 31;
  int n = lane & 15;
  int k0 = (lane >> 4) * 16;
  const bf16* p = base + n * stride + k0;
  v16bf f;
#pragma unroll
  for (int j = 0; j < 16; ++j) f[j] = p[j];
  return f;
}

// 2x16B vector load of a fragment from global
__device__ __forceinline__ v16bf load2x16B(const bf16* p0, const bf16* p1) {
  union { uint4 u[2]; v16bf v; } c;
  c.u[0] = *reinterpret_cast<const uint4*>(p0);
  c.u[1] = *reinterpret_cast<const uint4*>(p1);
  return c.v;
}

// B fragment from global k-contiguous storage
__device__ __forceinline__ v16bf frag_bT_g(const bf16* base, int stride) {
  int lane = threadIdx.x & 31;
  int n = lane & 15;
  int k0 = (lane >> 4) * 16;
  const bf16* p = base + (size_t)n * stride + k0;
  return load2x16B(p, p + 8);
}

__device__ __forceinline__ v8f wmma_bf16(v16bf a, v16bf b, v8f c) {
  return __builtin_amdgcn_wmma_f32_16x16x32_bf16(false, a, false, b,
                                                 (short)0, c, false, false);
}

__device__ __forceinline__ float elu_f(float v) {
  return v > 0.0f ? v : (__expf(v) - 1.0f);
}

// ---------------- generic bf16 GEMM (barrier-free, LDS-free, pipelined) -----
// A : MxK row-major bf16 (K multiple of 64; M multiple of 256)
// BT: [Npad64][K] bf16, k-contiguous, rows >= N zero-filled
// Block: 256 thr (8 waves). Tile 256x64; each wave owns 32 rows x 64 cols.
struct Frags { v16bf a0, a1, b[4]; };

__device__ __forceinline__ Frags load_frags(const bf16* ap0, const bf16* ap1,
                                            const bf16* const* bp, int k0) {
  Frags f;
  f.a0 = load2x16B(ap0 + k0, ap0 + k0 + 16);
  f.a1 = load2x16B(ap1 + k0, ap1 + k0 + 16);
#pragma unroll
  for (int nt = 0; nt < 4; ++nt)
    f.b[nt] = load2x16B(bp[nt] + k0, bp[nt] + k0 + 16);
  return f;
}

__device__ __forceinline__ void do_wmma(v8f acc[2][4], const Frags& f) {
#pragma unroll
  for (int nt = 0; nt < 4; ++nt) acc[0][nt] = wmma_bf16(f.a0, f.b[nt], acc[0][nt]);
#pragma unroll
  for (int nt = 0; nt < 4; ++nt) acc[1][nt] = wmma_bf16(f.a1, f.b[nt], acc[1][nt]);
}

__global__ __launch_bounds__(256) void gemm_bf16_kernel(
    const bf16* __restrict__ A, const bf16* __restrict__ BT,
    const float* __restrict__ bias, float* __restrict__ Cf,
    bf16* __restrict__ Cb, int M, int N, int K, int act)
{
  int tid  = threadIdx.x;
  int lane = tid & 31;
  int wave = tid >> 5;
  int mBase = blockIdx.y * 256 + wave * 32;   // this wave's 32 rows
  int nBase = blockIdx.x * 64;

  v8f acc[2][4];
#pragma unroll
  for (int mt = 0; mt < 2; ++mt)
#pragma unroll
    for (int nt = 0; nt < 4; ++nt) acc[mt][nt] = {};

  const bf16* ap0 = A + (size_t)(mBase + (lane & 15)) * K + (lane >> 4) * 8;
  const bf16* ap1 = ap0 + (size_t)16 * K;
  const bf16* bp[4];
#pragma unroll
  for (int nt = 0; nt < 4; ++nt)
    bp[nt] = BT + (size_t)(nBase + nt * 16 + (lane & 15)) * K + (lane >> 4) * 8;

  // 2-stage software pipeline over 64-wide K slices
  Frags s0 = load_frags(ap0, ap1, bp, 0);
  int k0 = 0;
  for (; k0 + 64 < K; k0 += 64) {
    Frags s1 = load_frags(ap0, ap1, bp, k0 + 32);
    do_wmma(acc, s0);
    s0 = load_frags(ap0, ap1, bp, k0 + 64);
    do_wmma(acc, s1);
  }
  Frags s1 = load_frags(ap0, ap1, bp, K - 32);
  do_wmma(acc, s0);
  do_wmma(acc, s1);

  int cl = lane & 15;
#pragma unroll
  for (int mt = 0; mt < 2; ++mt) {
    int rbase = mBase + mt * 16 + (lane >> 4) * 8;
#pragma unroll
    for (int nt = 0; nt < 4; ++nt) {
      int col = nBase + nt * 16 + cl;
      if (col < N) {
        float bv = bias ? bias[col] : 0.0f;
#pragma unroll
        for (int r = 0; r < 8; ++r) {
          float v = acc[mt][nt][r] + bv;
          if (act) v = elu_f(v);
          size_t o = (size_t)(rbase + r) * N + col;
          if (Cf) Cf[o] = v;
          if (Cb) Cb[o] = (bf16)v;
        }
      }
    }
  }
}

// ---------------- fp32 -> bf16 conversion -----------------------------------
__global__ void cvt_f32_bf16(const float* __restrict__ src,
                             bf16* __restrict__ dst, long n)
{
  long i = (long)blockIdx.x * blockDim.x + threadIdx.x;
  long stride = (long)gridDim.x * blockDim.x;
  for (; i < n; i += stride) dst[i] = (bf16)src[i];
}

// ---------------- fp32 [R][C] -> bf16 transposed [Cpad][R] ------------------
__global__ void transpose_cvt(const float* __restrict__ src,
                              bf16* __restrict__ dst, int R, int C, int Cpad)
{
  long total = (long)Cpad * R;
  long i = (long)blockIdx.x * blockDim.x + threadIdx.x;
  long stride = (long)gridDim.x * blockDim.x;
  for (; i < total; i += stride) {
    int c = (int)(i / R);
    int r = (int)(i - (long)c * R);
    dst[i] = (c < C) ? (bf16)src[(size_t)r * C + c] : (bf16)0.0f;
  }
}

// ---------------- reparameterization: z = mean + exp(0.5 lnvar) * eps -------
__global__ void reparam_kernel(const float* __restrict__ mean,
                               const float* __restrict__ lnvar,
                               const float* __restrict__ eps,
                               float* __restrict__ z, bf16* __restrict__ zb,
                               int n)
{
  int i = blockIdx.x * blockDim.x + threadIdx.x;
  int stride = gridDim.x * blockDim.x;
  for (; i < n; i += stride) {
    float v = mean[i] + __expf(0.5f * lnvar[i]) * eps[i];
    z[i] = v;
    if (zb) zb[i] = (bf16)v;
  }
}

// ---------------- Hamiltonian integrator ------------------------------------
// H MLP: [40 -> 256 (elu) -> 256 (elu) -> 1]
// g = d/dyy H(concat(yy, z_phy)); dyydt = [g[:,16:32], -g[:,0:16]]
// Per WG: 32 samples, 510 sequential steps.  B-fragments loaded once per
// column tile (nt) and reused across both M-subtiles.
#define HAM_SMEM_BYTES 272384
__global__ __launch_bounds__(256) void ham_kernel(
    const float* __restrict__ HW1, const float* __restrict__ Hb1,
    const bf16* __restrict__ W2tb, const float* __restrict__ Hb2,
    const float* __restrict__ HW3, const bf16* __restrict__ W2b,
    const float* __restrict__ init_yy, const float* __restrict__ z_phy,
    float* __restrict__ y_seq)
{
  extern __shared__ char smem[];
  bf16*  sW2t = (bf16*)smem;                // [256][264]  W2t[h2][h1]
  bf16*  sW1t = sW2t + 256 * 264;           // [256][72]   W1t[h][i] (i 40..63 = 0)
  bf16*  sW1b = sW1t + 256 * 72;            // [32][264]   W1[i][h], i<32
  bf16*  sIn  = sW1b + 32 * 264;            // [32][72]    cols 0..31 yy, 32..39 z_phy
  bf16*  sH1  = sIn  + 32 * 72;             // [32][264]   elu(pre1)
  bf16*  sE1  = sH1  + 32 * 264;            // [32][264]   elu'(pre1)
  bf16*  sD2  = sE1  + 32 * 264;            // [32][264]   dh2
  bf16*  sD1  = sD2  + 32 * 264;            // [32][264]   dh1
  float* sW3  = (float*)(sD1 + 32 * 264);   // [256]
  float* sb1  = sW3 + 256;
  float* sb2  = sb1 + 256;
  float* sYY  = sb2 + 256;                  // [32][32]
  float* sG   = sYY + 32 * 32;              // [32][32]

  int tid  = threadIdx.x;
  int lane = tid & 31;
  int wave = tid >> 5;
  int sBase = blockIdx.x * 32;

  // --- one-time LDS fills ---
  for (int i = tid; i < 256 * 32; i += 256) {       // W2t rows: coalesced 16B
    int r = i >> 5, ch = (i & 31) * 8;
    *reinterpret_cast<uint4*>(&sW2t[r * 264 + ch]) =
        *reinterpret_cast<const uint4*>(&W2tb[r * 256 + ch]);
  }
  for (int i = tid; i < 256 * 64; i += 256) {       // W1t[h][i]
    int h = i >> 6, c = i & 63;
    sW1t[h * 72 + c] = (c < 40) ? (bf16)HW1[c * 256 + h] : (bf16)0.0f;
  }
  for (int i = tid; i < 32 * 256; i += 256) {       // W1[i][h], i<32
    int r = i >> 8, c = i & 255;
    sW1b[r * 264 + c] = (bf16)HW1[r * 256 + c];
  }
  for (int i = tid; i < 256; i += 256) {
    sW3[i] = HW3[i];  sb1[i] = Hb1[i];  sb2[i] = Hb2[i];
  }
  for (int i = tid; i < 32 * 72; i += 256) sIn[i] = (bf16)0.0f;
  for (int i = tid; i < 32 * 32; i += 256)
    sYY[i] = init_yy[(size_t)sBase * 32 + i];
  __syncthreads();
  for (int i = tid; i < 32 * 8; i += 256) {
    int s = i >> 3, c = i & 7;
    sIn[s * 72 + 32 + c] = (bf16)z_phy[(sBase + s) * 8 + c];
  }
  for (int i = tid; i < 32 * 16; i += 256) {        // y_seq[t=0]
    int s = i >> 4, y = i & 15;
    y_seq[(size_t)(sBase + s) * 16 + y] = sYY[s * 32 + y];
  }
  __syncthreads();

  const float dt = 0.05f;  // DT / INTG_LEV
  int cl = lane & 15;
  int rsel = (lane >> 4) * 8;

  for (int step = 1; step <= 510; ++step) {
    for (int i = tid; i < 32 * 32; i += 256) {      // refresh yy cols (bf16)
      int s = i >> 5, c = i & 31;
      sIn[s * 72 + c] = (bf16)sYY[i];
    }
    __syncthreads();

    // GEMM1: pre1 = In(32x64) @ W1(64x256) + b1 -> h1, e1
#pragma unroll
    for (int j = 0; j < 2; ++j) {
      int nt = wave * 2 + j;
      v16bf b0 = frag_bT(&sW1t[(nt * 16) * 72 + 0], 72);
      v16bf b1 = frag_bT(&sW1t[(nt * 16) * 72 + 32], 72);
      int col = nt * 16 + cl;
      float bv = sb1[col];
#pragma unroll
      for (int mt = 0; mt < 2; ++mt) {
        v8f acc = {};
        acc = wmma_bf16(frag_a(&sIn[mt * 16 * 72 + 0], 72),  b0, acc);
        acc = wmma_bf16(frag_a(&sIn[mt * 16 * 72 + 32], 72), b1, acc);
        int rb = mt * 16 + rsel;
#pragma unroll
        for (int r = 0; r < 8; ++r) {
          float pre = acc[r] + bv;
          float h = pre > 0.0f ? pre : (__expf(pre) - 1.0f);
          float e = pre > 0.0f ? 1.0f : (h + 1.0f); // elu'(x)=exp(x)=elu(x)+1
          sH1[(rb + r) * 264 + col] = (bf16)h;
          sE1[(rb + r) * 264 + col] = (bf16)e;
        }
      }
    }
    __syncthreads();

    // GEMM2: pre2 = h1 @ W2 + b2 ; dh2 = elu'(pre2) * W3
#pragma unroll
    for (int j = 0; j < 2; ++j) {
      int nt = wave * 2 + j;
      v16bf bf[8];
#pragma unroll
      for (int kk = 0; kk < 8; ++kk)
        bf[kk] = frag_bT(&sW2t[(nt * 16) * 264 + kk * 32], 264);
      int col = nt * 16 + cl;
      float bv = sb2[col], w3 = sW3[col];
#pragma unroll
      for (int mt = 0; mt < 2; ++mt) {
        v8f acc = {};
#pragma unroll
        for (int kk = 0; kk < 8; ++kk)
          acc = wmma_bf16(frag_a(&sH1[mt * 16 * 264 + kk * 32], 264), bf[kk], acc);
        int rb = mt * 16 + rsel;
#pragma unroll
        for (int r = 0; r < 8; ++r) {
          float pre = acc[r] + bv;
          float e = pre > 0.0f ? 1.0f : __expf(pre);
          sD2[(rb + r) * 264 + col] = (bf16)(e * w3);
        }
      }
    }
    __syncthreads();

    // GEMM3: dh1 = (dh2 @ W2^T) * e1 ; B streamed from L2 (W2b, [h1][h2])
#pragma unroll
    for (int j = 0; j < 2; ++j) {
      int nt = wave * 2 + j;
      v16bf bf[8];
#pragma unroll
      for (int kk = 0; kk < 8; ++kk)
        bf[kk] = frag_bT_g(W2b + (size_t)(nt * 16) * 256 + kk * 32, 256);
      int col = nt * 16 + cl;
#pragma unroll
      for (int mt = 0; mt < 2; ++mt) {
        v8f acc = {};
#pragma unroll
        for (int kk = 0; kk < 8; ++kk)
          acc = wmma_bf16(frag_a(&sD2[mt * 16 * 264 + kk * 32], 264), bf[kk], acc);
        int rb = mt * 16 + rsel;
#pragma unroll
        for (int r = 0; r < 8; ++r) {
          float v = acc[r] * (float)sE1[(rb + r) * 264 + col];
          sD1[(rb + r) * 264 + col] = (bf16)v;
        }
      }
    }
    __syncthreads();

    // GEMM4: g = dh1 @ W1^T (first 32 input cols) -> waves 0..3
    if (wave < 4) {
      int mt = wave & 1, nt = wave >> 1;
      v8f acc = {};
#pragma unroll
      for (int kk = 0; kk < 8; ++kk) {
        v16bf a = frag_a(&sD1[mt * 16 * 264 + kk * 32], 264);
        v16bf b = frag_bT(&sW1b[(nt * 16) * 264 + kk * 32], 264);
        acc = wmma_bf16(a, b, acc);
      }
      int col = nt * 16 + cl, rb = mt * 16 + rsel;
#pragma unroll
      for (int r = 0; r < 8; ++r) sG[(rb + r) * 32 + col] = acc[r];
    }
    __syncthreads();

    // symplectic update + strided emission
    bool emit = (step & 1) == 0;
    int tOut = step >> 1;
    for (int i = tid; i < 32 * 32; i += 256) {
      int s = i >> 5, c = i & 31;
      float d = (c < 16) ? sG[s * 32 + 16 + c] : -sG[s * 32 + (c - 16)];
      float ny = sYY[i] + dt * d;
      sYY[i] = ny;
      if (emit && c < 16)
        y_seq[((size_t)tOut * 512 + (sBase + s)) * 16 + c] = ny;
    }
    __syncthreads();
  }
}

// ---------------- final: x_mean = einsum(y_seq, map_W) + map_b + x_res ------
__global__ __launch_bounds__(256) void final_kernel(
    const float* __restrict__ y_seq, const float* __restrict__ map_W,
    const float* __restrict__ map_b, const float* __restrict__ x_res,
    float* __restrict__ x_mean, float* __restrict__ x_lnvar)
{
  __shared__ float sW[16 * 32];
  __shared__ float sB[32];
  int t = threadIdx.x;          // time index 0..255
  int n = blockIdx.x;           // sample
  for (int i = t; i < 16 * 32; i += 256) sW[i] = map_W[i];
  if (t < 32) sB[t] = map_b[t];
  __syncthreads();

  float y[16];
  const float* yp = y_seq + ((size_t)t * 512 + n) * 16;
#pragma unroll
  for (int j = 0; j < 16; ++j) y[j] = yp[j];
#pragma unroll
  for (int x = 0; x < 32; ++x) {
    float acc = sB[x];
#pragma unroll
    for (int j = 0; j < 16; ++j) acc += y[j] * sW[j * 32 + x];
    size_t o = ((size_t)n * 32 + x) * 256 + t;
    x_mean[o] = acc + x_res[o];
  }
  if (n == 0 && t == 0) x_lnvar[0] = -2.0f;
}

// ---------------------------------------------------------------------------
extern "C" void kernel_launch(void* const* d_in, const int* in_sizes, int n_in,
                              void* d_out, int out_size, void* d_ws, size_t ws_size,
                              hipStream_t stream)
{
  (void)in_sizes; (void)n_in; (void)out_size; (void)ws_size;
  const float* x        = (const float*)d_in[0];
  const float* eps_phy  = (const float*)d_in[1];
  const float* eps_aux2 = (const float*)d_in[2];
  const float* fW0 = (const float*)d_in[3];  const float* fb0 = (const float*)d_in[4];
  const float* fW1 = (const float*)d_in[5];  const float* fb1 = (const float*)d_in[6];
  const float* HW1 = (const float*)d_in[27]; const float* Hb1 = (const float*)d_in[28];
  const float* HW2 = (const float*)d_in[29]; const float* Hb2 = (const float*)d_in[30];
  const float* HW3 = (const float*)d_in[31];
  const float* mapW = (const float*)d_in[33]; const float* mapB = (const float*)d_in[34];
  const float* aW0 = (const float*)d_in[35]; const float* ab0 = (const float*)d_in[36];
  const float* aW1 = (const float*)d_in[37]; const float* ab1 = (const float*)d_in[38];

  float* out = (float*)d_out;
  float* o_zpm = out;            // z_phy_mean   512x8
  float* o_zpl = out + 4096;     // z_phy_lnvar  512x8
  float* o_zam = out + 8192;     // z_aux2_mean  512x64
  float* o_zal = out + 40960;    // z_aux2_lnvar 512x64
  float* o_xm  = out + 73728;    // x_mean       512x32x256
  float* o_xlv = out + 4268032;  // x_lnvar      1

  // ---- workspace carve ----
  char* p = (char*)d_ws;
  auto alloc = [&](size_t bytes) {
    char* r = p; p += (bytes + 255) & ~(size_t)255; return r;
  };
  bf16* xb    = (bf16*)alloc(4194304ull * 2);          // x bf16 (A)
  bf16* wf1T  = (bf16*)alloc(1024ull * 8192 * 2);      // feat W0^T [1024][8192]
  bf16* wf2T  = (bf16*)alloc(1024ull * 1024 * 2);      // feat W1^T [1024][1024]
  bf16* hf1b  = (bf16*)alloc(524288ull * 2);           // feat hidden (A)
  bf16* featb = (bf16*)alloc(524288ull * 2);           // feat (A)
  bf16* headW0T[5]; bf16* headW1T[5]; bf16* headHb[5];
  const int headW0idx[5] = {7, 11, 15, 19, 23};
  const int headNout[5]  = {8, 8, 64, 64, 32};
  for (int h = 0; h < 5; ++h) {
    headW0T[h] = (bf16*)alloc(256ull * 1024 * 2);      // [256][1024]
    headW1T[h] = (bf16*)alloc(64ull * 256 * 2);        // [64][256] zero-padded
    headHb[h]  = (bf16*)alloc(131072ull * 2);          // 512x256 hidden (A)
  }
  float* init_yy_ws = (float*)alloc(16384ull * 4);     // 512x32
  float* z_phy_ws   = (float*)alloc(4096ull * 4);
  float* z_aux2_ws  = (float*)alloc(32768ull * 4);
  bf16*  z_aux2_b   = (bf16*)alloc(32768ull * 2);
  bf16*  aW0T       = (bf16*)alloc(256ull * 64 * 2);   // [256][64]
  bf16*  aW1T       = (bf16*)alloc(8192ull * 256 * 2); // [8192][256]
  bf16*  aHb        = (bf16*)alloc(131072ull * 2);     // 512x256 (A)
  bf16*  W2b        = (bf16*)alloc(65536ull * 2);      // H W2 bf16 [h1][h2]
  bf16*  W2tb       = (bf16*)alloc(65536ull * 2);      // H W2^T bf16 [h2][h1]
  float* x_res      = (float*)alloc(4194304ull * 4);
  float* y_seq      = (float*)alloc(2097152ull * 4);   // 256x512x16

  auto cvt = [&](const float* s, bf16* d, long n) {
    int blocks = (int)std::min((n + 255) / 256, (long)2048);
    cvt_f32_bf16<<<dim3(blocks), dim3(256), 0, stream>>>(s, d, n);
  };
  auto tcvt = [&](const float* s, bf16* d, int R, int C, int Cpad) {
    long n = (long)Cpad * R;
    int blocks = (int)std::min((n + 255) / 256, (long)2048);
    transpose_cvt<<<dim3(blocks), dim3(256), 0, stream>>>(s, d, R, C, Cpad);
  };
  auto gemm = [&](const bf16* A, const bf16* BT, const float* bias,
                  float* Cf, bf16* Cb, int M, int N, int K, int act) {
    dim3 grid((N + 63) / 64, (M + 255) / 256);
    gemm_bf16_kernel<<<grid, dim3(256), 0, stream>>>(A, BT, bias, Cf, Cb, M, N, K, act);
  };

  // ---- weight prep (bf16, k-contiguous) ----
  cvt(x, xb, 4194304);
  tcvt(fW0, wf1T, 8192, 1024, 1024);
  tcvt(fW1, wf2T, 1024, 1024, 1024);
  for (int h = 0; h < 5; ++h) {
    tcvt((const float*)d_in[headW0idx[h]],     headW0T[h], 1024, 256, 256);
    tcvt((const float*)d_in[headW0idx[h] + 2], headW1T[h], 256, headNout[h], 64);
  }
  tcvt(aW0, aW0T, 64, 256, 256);
  tcvt(aW1, aW1T, 256, 8192, 8192);
  cvt(HW2, W2b, 65536);
  tcvt(HW2, W2tb, 256, 256, 256);

  // ---- feature extractor (elu on both layers) ----
  gemm(xb,   wf1T, fb0, nullptr, hf1b,  512, 1024, 8192, 1);
  gemm(hf1b, wf2T, fb1, nullptr, featb, 512, 1024, 1024, 1);

  // ---- heads ----
  float* headDst[5] = {o_zpm, o_zpl, o_zam, o_zal, init_yy_ws};
  for (int h = 0; h < 5; ++h) {
    const float* b0 = (const float*)d_in[headW0idx[h] + 1];
    const float* b1 = (const float*)d_in[headW0idx[h] + 3];
    gemm(featb, headW0T[h], b0, nullptr, headHb[h], 512, 256, 1024, 1);
    gemm(headHb[h], headW1T[h], b1, headDst[h], nullptr, 512, headNout[h], 256, 0);
  }

  // ---- reparameterization ----
  reparam_kernel<<<dim3(16),  dim3(256), 0, stream>>>(o_zpm, o_zpl, eps_phy,
                                                      z_phy_ws, (bf16*)nullptr, 4096);
  reparam_kernel<<<dim3(128), dim3(256), 0, stream>>>(o_zam, o_zal, eps_aux2,
                                                      z_aux2_ws, z_aux2_b, 32768);

  // ---- aux2 residual decoder ----
  gemm(z_aux2_b, aW0T, ab0, nullptr, aHb,   512, 256,  64,  1);
  gemm(aHb,      aW1T, ab1, x_res,  nullptr, 512, 8192, 256, 0);

  // ---- Hamiltonian integration (persistent scan, 16 WGs x 32 samples) ----
  hipFuncSetAttribute((const void*)ham_kernel,
                      hipFuncAttributeMaxDynamicSharedMemorySize, HAM_SMEM_BYTES);
  ham_kernel<<<dim3(16), dim3(256), HAM_SMEM_BYTES, stream>>>(
      HW1, Hb1, W2tb, Hb2, HW3, W2b, init_yy_ws, z_phy_ws, y_seq);

  // ---- final assembly ----
  final_kernel<<<dim3(512), dim3(256), 0, stream>>>(y_seq, mapW, mapB, x_res,
                                                    o_xm, o_xlv);
}
